// S4DAttention_53025666236919
// MI455X (gfx1250) — matrix operand order repaired and checked by reading, not compiled, not run on hardware
//
#include <hip/hip_runtime.h>
#include <hip/hip_bf16.h>
#include <stdint.h>
#include <math.h>

// ---------------------------------------------------------------------------
// Problem constants (fixed by the reference): B=8, L=2048, H=768, N=32
// ---------------------------------------------------------------------------
#define HDIM   768
#define BATCH  8
#define SEQ    2048
#define NSTATE 32
#define MROWS  (BATCH * SEQ)   // 16384 rows for all GEMMs

// LDS staging geometry for the GEMM (per 128x256 block tile, K-chunk = 32)
#define ROWH      40                 // 32 bf16 + 8 pad halves = 80 B row stride
#define LDSA_ROWS 128
#define LDSB_ROWS 256
#define KCHUNKS   (HDIM / 32)        // 24

typedef __attribute__((ext_vector_type(16))) __bf16 v16bf;
typedef __attribute__((ext_vector_type(8)))  float  v8f;

// Builtin signature (from compiler diagnostic): pointers to 16B int vectors,
// global operand in AS(1), LDS operand in AS(3).
typedef int v4i_vec __attribute__((vector_size(16)));
typedef __attribute__((address_space(1))) v4i_vec gv4i;
typedef __attribute__((address_space(3))) v4i_vec lv4i;

#if defined(__has_builtin)
#  if __has_builtin(__builtin_amdgcn_global_load_async_to_lds_b128)
#    define ASYNC_LDS 1
#  endif
#endif
#ifndef ASYNC_LDS
#  define ASYNC_LDS 0
#endif

// ----------------------------- helpers -------------------------------------
static __device__ __forceinline__ uint16_t f2bf(float f) {
  union { float f; uint32_t u; } v; v.f = f;
  uint32_t lsb = (v.u >> 16) & 1u;
  uint32_t r = v.u + 0x7fffu + lsb;       // round-to-nearest-even
  return (uint16_t)(r >> 16);
}

static __device__ __forceinline__ float gelu_tanh(float x) {
  float x3 = x * x * x;
  float t = tanhf(0.7978845608028654f * (x + 0.044715f * x3));
  return 0.5f * x * (1.0f + t);
}

// 16-byte async copy global -> LDS (falls back to load + ds_store)
static __device__ __forceinline__ void cp16(const uint16_t* g, uint16_t* l) {
#if ASYNC_LDS
  __builtin_amdgcn_global_load_async_to_lds_b128(
      (gv4i*)(uintptr_t)g,                 // generic -> AS(1) via integer cast
      (lv4i*)(uint32_t)(uintptr_t)l,       // flat -> LDS offset = addr[31:0]
      0, 0);
#else
  *(uint4*)l = *(const uint4*)g;
#endif
}

static __device__ __forceinline__ void wait_async_chunk() {
#if ASYNC_LDS
#  if __has_builtin(__builtin_amdgcn_s_wait_asynccnt)
  __builtin_amdgcn_s_wait_asynccnt(6);     // allow next chunk's 6 copies in flight
#  else
  asm volatile("s_wait_asynccnt 6" ::: "memory");
#  endif
#endif
}

static __device__ __forceinline__ void wait_async_all() {
#if ASYNC_LDS
#  if __has_builtin(__builtin_amdgcn_s_wait_asynccnt)
  __builtin_amdgcn_s_wait_asynccnt(0);
#  else
  asm volatile("s_wait_asynccnt 0" ::: "memory");
#  endif
#endif
}

// ----------------------- elementwise conversion kernels ---------------------
__global__ void f32_to_bf16_kernel(const float* __restrict__ in,
                                   uint16_t* __restrict__ out, int n) {
  int i = blockIdx.x * blockDim.x + threadIdx.x;
  if (i < n) out[i] = f2bf(in[i]);
}

// Wt[n*H + k] = bf16(W[k*H + n])  (transpose so B loads run along K)
__global__ void wt_convert_kernel(const float* __restrict__ W,
                                  uint16_t* __restrict__ Wt) {
  int i = blockIdx.x * blockDim.x + threadIdx.x;
  if (i >= HDIM * HDIM) return;
  int n = i / HDIM, k = i % HDIM;
  Wt[i] = f2bf(W[k * HDIM + n]);
}

__global__ void mul_to_bf16_kernel(const float* __restrict__ a,
                                   const float* __restrict__ b,
                                   uint16_t* __restrict__ out, int n) {
  int i = blockIdx.x * blockDim.x + threadIdx.x;
  if (i < n) out[i] = f2bf(a[i] * b[i]);
}

// ----------------------------- WMMA GEMM ------------------------------------
// Y(f32, MxN) = epi( A(bf16, MxK row-major) x Wt(bf16, NxK)^T + bias )
// Block: 256 threads = 8 waves; block tile 128x256; wave tile 64x64.
// A/B K-chunks are double-buffered in LDS via async global->LDS copies
// (ASYNCcnt + workgroup barrier pipeline), fragments read with ds_load_b128.
// EPI: 0 = bias only, 1 = bias + gelu
template <int EPI>
__global__ __launch_bounds__(256) void gemm_wmma_kernel(
    const uint16_t* __restrict__ A,
    const uint16_t* __restrict__ Wt,
    const float* __restrict__ bias,
    float* __restrict__ Y) {
  __shared__ __align__(16) uint16_t ldsA[2 * LDSA_ROWS * ROWH];  // 2 x 10 KB
  __shared__ __align__(16) uint16_t ldsB[2 * LDSB_ROWS * ROWH];  // 2 x 20 KB

  const int K = HDIM, Nn = HDIM;
  const int lane   = threadIdx.x & 31;
  const int wid    = threadIdx.x >> 5;
  const int waveM  = wid >> 2;          // 2 waves along M
  const int waveN  = wid & 3;           // 4 waves along N
  const int mBase0 = blockIdx.y * 128;
  const int nBase0 = blockIdx.x * 256;
  const int mBase  = mBase0 + waveM * 64;
  const int nBase  = nBase0 + waveN * 64;
  const int lm     = lane & 15;
  const int hiHalf = lane >> 4;         // lanes 16..31 hold the other K half
  const int cA     = hiHalf * 8;        // A-matrix lane-half K offset (halves)

  // ---- async fill addressing: thread t copies 16B; row = t/4, koff = (t%4)*8
  const int fm = threadIdx.x >> 2;              // 0..63
  const int fk = (threadIdx.x & 3) * 8;         // half offset within chunk
  const uint16_t* gA = A  + (size_t)(mBase0 + fm) * K + fk;
  const uint16_t* gB = Wt + (size_t)(nBase0 + fm) * K + fk;

  auto issue = [&](int c) {
    const int buf = c & 1;
    const int k0h = c * 32;
    uint16_t* lA = ldsA + buf * (LDSA_ROWS * ROWH) + fm * ROWH + fk;
    uint16_t* lB = ldsB + buf * (LDSB_ROWS * ROWH) + fm * ROWH + fk;
#pragma unroll
    for (int j = 0; j < 2; ++j)   // 128 A rows
      cp16(gA + (size_t)j * 64 * K + k0h, lA + j * 64 * ROWH);
#pragma unroll
    for (int j = 0; j < 4; ++j)   // 256 B rows
      cp16(gB + (size_t)j * 64 * K + k0h, lB + j * 64 * ROWH);
  };

  v8f acc[4][4];
#pragma unroll
  for (int mi = 0; mi < 4; ++mi)
#pragma unroll
    for (int ni = 0; ni < 4; ++ni)
#pragma unroll
      for (int e = 0; e < 8; ++e) acc[mi][ni][e] = 0.0f;

  issue(0);
  issue(1);

  for (int c = 0; c < KCHUNKS; ++c) {
    if (c == KCHUNKS - 1) wait_async_all();
    else                  wait_async_chunk();
    __syncthreads();                       // chunk c resident in buf (c&1)

    const uint16_t* bufA = ldsA + (c & 1) * (LDSA_ROWS * ROWH);
    const uint16_t* bufB = ldsB + (c & 1) * (LDSB_ROWS * ROWH);

    union { v16bf v; uint4 q[2]; } af[4], bf[4];
#pragma unroll
    for (int mi = 0; mi < 4; ++mi) {
      const uint16_t* p = bufA + (waveM * 64 + mi * 16 + lm) * ROWH + cA;
      af[mi].q[0] = *(const uint4*)p;          // K = cA .. cA+7
      af[mi].q[1] = *(const uint4*)(p + 16);   // K = 16+cA .. 16+cA+7
    }
#pragma unroll
    for (int ni = 0; ni < 4; ++ni) {
      const uint16_t* p = bufB + (waveN * 64 + ni * 16 + lm) * ROWH + hiHalf * 8;
      bf[ni].q[0] = *(const uint4*)p;          // K = h*8 .. +7
      bf[ni].q[1] = *(const uint4*)(p + 16);   // K = 16+h*8 .. +7
    }

#pragma unroll
    for (int mi = 0; mi < 4; ++mi)
#pragma unroll
      for (int ni = 0; ni < 4; ++ni)
        acc[mi][ni] = __builtin_amdgcn_wmma_f32_16x16x32_bf16(
            false, af[mi].v, false, bf[ni].v, (short)0, acc[mi][ni], false, false);

    __syncthreads();                       // all waves done reading buf (c&1)
    if (c + 2 < KCHUNKS) issue(c + 2);     // refill the buffer just released
  }

  // Epilogue: C/D layout — element e of v8f is M = e + hiHalf*8, N = lane%16
#pragma unroll
  for (int ni = 0; ni < 4; ++ni) {
    int n = nBase + ni * 16 + lm;
    float bv = bias[n];
#pragma unroll
    for (int mi = 0; mi < 4; ++mi) {
#pragma unroll
      for (int e = 0; e < 8; ++e) {
        int m = mBase + mi * 16 + hiHalf * 8 + e;
        float val = acc[mi][ni][e] + bv;
        if (EPI == 1) val = gelu_tanh(val);
        Y[(size_t)m * Nn + n] = val;
      }
    }
  }
}

// ----------------------------- S4D scan -------------------------------------
// One wave per (batch, channel); lane = diagonal mode n (N=32 == wave32).
// s[l] = Abar*s[l±1] + x[l];  y[l] = 2*Re(sum_n Cbar_n s_n[l]) + D*x[l]
// DIR=+1 forward scan, DIR=-1 backward (time-reversed) scan — replaces the
// reference's flip->conv->flip with zero data movement.
template <int DIR>
__global__ __launch_bounds__(256) void s4d_scan_kernel(
    const float* __restrict__ x,            // (B, L, H) f32
    const float* __restrict__ log_dt,       // (H)
    const float* __restrict__ log_A_real,   // (H, N)
    const float* __restrict__ A_imag,       // (H, N)
    const float* __restrict__ C_re,         // (H, N)
    const float* __restrict__ C_im,         // (H, N)
    const float* __restrict__ Dvec,         // (H)
    uint16_t* __restrict__ out) {           // (B, L, H) bf16
  const int lane = threadIdx.x & 31;
  const int wid  = blockIdx.x * (blockDim.x >> 5) + (threadIdx.x >> 5);
  const int b = wid / HDIM;
  const int h = wid % HDIM;

  // Per-mode discretized parameters (ZOH, B=1)
  const float dt  = expf(log_dt[h]);
  const int   pi  = h * NSTATE + lane;
  const float Ar  = -expf(log_A_real[pi]);
  const float Ai  = A_imag[pi];
  const float dAr = dt * Ar, dAi = dt * Ai;
  const float ex  = expf(dAr);
  const float Abr = ex * cosf(dAi);       // Abar = exp(dt*A)
  const float Abi = ex * sinf(dAi);
  const float Er  = Abr - 1.0f, Ei = Abi; // exp(dt*A) - 1
  const float cr  = C_re[pi], ci = C_im[pi];
  const float tr  = cr * Er - ci * Ei;    // C * (exp(dtA)-1)
  const float ti  = cr * Ei + ci * Er;
  const float inv = 1.0f / (Ar * Ar + Ai * Ai);
  const float Cbr = (tr * Ar + ti * Ai) * inv;   // ... / A
  const float Cbi = (ti * Ar - tr * Ai) * inv;
  const float Dh  = Dvec[h];

  float sr = 0.0f, si = 0.0f;
  const size_t base = (size_t)b * SEQ * HDIM + h;
  for (int t = 0; t < SEQ; ++t) {
    const int l = (DIR > 0) ? t : (SEQ - 1 - t);
    const float uv = x[base + (size_t)l * HDIM];
    const float nsr = fmaf(Abr, sr, fmaf(-Abi, si, uv));
    const float nsi = fmaf(Abr, si, Abi * sr);
    sr = nsr; si = nsi;
    float c = Cbr * sr - Cbi * si;
    // wave32 butterfly reduction over the 32 modes
    c += __shfl_xor(c, 16, 32);
    c += __shfl_xor(c, 8, 32);
    c += __shfl_xor(c, 4, 32);
    c += __shfl_xor(c, 2, 32);
    c += __shfl_xor(c, 1, 32);
    if (lane == 0) out[base + (size_t)l * HDIM] = f2bf(2.0f * c + Dh * uv);
  }
}

// ----------------------------- launcher -------------------------------------
extern "C" void kernel_launch(void* const* d_in, const int* in_sizes, int n_in,
                              void* d_out, int out_size, void* d_ws, size_t ws_size,
                              hipStream_t stream) {
  (void)in_sizes; (void)n_in; (void)out_size; (void)ws_size;
  const float* u        = (const float*)d_in[0];
  const float* W_pf     = (const float*)d_in[1];
  const float* b_pf     = (const float*)d_in[2];
  const float* W_pb     = (const float*)d_in[3];
  const float* b_pb     = (const float*)d_in[4];
  const float* W_postf  = (const float*)d_in[5];
  const float* b_postf  = (const float*)d_in[6];
  const float* W_postb  = (const float*)d_in[7];
  const float* b_postb  = (const float*)d_in[8];
  const float* W_out    = (const float*)d_in[9];
  const float* b_out    = (const float*)d_in[10];
  const float* f_log_dt = (const float*)d_in[11];
  const float* f_log_Ar = (const float*)d_in[12];
  const float* f_A_im   = (const float*)d_in[13];
  const float* f_C_re   = (const float*)d_in[14];
  const float* f_C_im   = (const float*)d_in[15];
  const float* f_D      = (const float*)d_in[16];
  const float* b_log_dt = (const float*)d_in[17];
  const float* b_log_Ar = (const float*)d_in[18];
  const float* b_A_im   = (const float*)d_in[19];
  const float* b_C_re   = (const float*)d_in[20];
  const float* b_C_im   = (const float*)d_in[21];
  const float* b_D      = (const float*)d_in[22];
  float* out = (float*)d_out;

  // Workspace layout (all 256B aligned)
  char* ws = (char*)d_ws;
  size_t off = 0;
  auto carve = [&](size_t bytes) -> void* {
    void* p = ws + off;
    off = (off + bytes + 255) & ~(size_t)255;
    return p;
  };
  const size_t NEL = (size_t)MROWS * HDIM;          // 12.58M elements
  uint16_t* u_bf = (uint16_t*)carve(NEL * 2);
  uint16_t* Wt[5];
  for (int i = 0; i < 5; ++i) Wt[i] = (uint16_t*)carve((size_t)HDIM * HDIM * 2);
  float*    xf   = (float*)   carve(NEL * 4);       // fwd act, later reused as yf
  float*    xb   = (float*)   carve(NEL * 4);       // bwd act, later reused as yb
  uint16_t* sf   = (uint16_t*)carve(NEL * 2);       // fwd S4D out, later p=yf*yb
  uint16_t* sb   = (uint16_t*)carve(NEL * 2);       // bwd S4D out

  const int nel = (int)NEL;
  const int cvt_blocks = (nel + 255) / 256;
  const int wgrid = (HDIM * HDIM + 255) / 256;
  dim3 ggrid(HDIM / 256, MROWS / 128);              // (3, 128)

  // Stage 0: precision conversion (u and all 5 weights -> bf16; W transposed)
  f32_to_bf16_kernel<<<cvt_blocks, 256, 0, stream>>>(u, u_bf, nel);
  wt_convert_kernel<<<wgrid, 256, 0, stream>>>(W_pf,    Wt[0]);
  wt_convert_kernel<<<wgrid, 256, 0, stream>>>(W_pb,    Wt[1]);
  wt_convert_kernel<<<wgrid, 256, 0, stream>>>(W_postf, Wt[2]);
  wt_convert_kernel<<<wgrid, 256, 0, stream>>>(W_postb, Wt[3]);
  wt_convert_kernel<<<wgrid, 256, 0, stream>>>(W_out,   Wt[4]);

  // Stage 1: input projections + GELU (flip commutes with pointwise ops)
  gemm_wmma_kernel<1><<<ggrid, 256, 0, stream>>>(u_bf, Wt[0], b_pf, xf);
  gemm_wmma_kernel<1><<<ggrid, 256, 0, stream>>>(u_bf, Wt[1], b_pb, xb);

  // Stage 2: S4D as diagonal recurrences (fwd scan + time-reversed scan)
  s4d_scan_kernel<+1><<<(BATCH * HDIM) / 8, 256, 0, stream>>>(
      xf, f_log_dt, f_log_Ar, f_A_im, f_C_re, f_C_im, f_D, sf);
  s4d_scan_kernel<-1><<<(BATCH * HDIM) / 8, 256, 0, stream>>>(
      xb, b_log_dt, b_log_Ar, b_A_im, b_C_re, b_C_im, b_D, sb);

  // Stage 3: post projections (reuse xf/xb buffers as yf/yb)
  gemm_wmma_kernel<0><<<ggrid, 256, 0, stream>>>(sf, Wt[2], b_postf, xf);
  gemm_wmma_kernel<0><<<ggrid, 256, 0, stream>>>(sb, Wt[3], b_postb, xb);

  // Stage 4: gate product (reuse sf as bf16 product buffer) + output projection
  mul_to_bf16_kernel<<<cvt_blocks, 256, 0, stream>>>(xf, xb, sf, nel);
  gemm_wmma_kernel<0><<<ggrid, 256, 0, stream>>>(sf, Wt[4], b_out, out);
}
